// Attention_16200616640529
// MI455X (gfx1250) — compile-verified, matrix operand order checked by reading, not run on hardware
//
#include <hip/hip_runtime.h>
#include <hip/hip_bf16.h>

typedef __attribute__((ext_vector_type(16))) _Float16 v16h;
typedef __attribute__((ext_vector_type(8)))  _Float16 h8;
typedef __attribute__((ext_vector_type(8)))  float    v8f;

#define WMMA_F16(A, B, C) \
  __builtin_amdgcn_wmma_f32_16x16x32_f16(false, (A), false, (B), (short)0, (C), false, false)

// ---------------------------------------------------------------------------
// Problem constants
// ---------------------------------------------------------------------------
#define B   8
#define C   256
#define HW  1024
#define GRP 8
#define NH  4
#define DH  64
#define OC3 (3 * C)   // 768

// ---------------------------------------------------------------------------
// CDNA5 wave32 WMMA fragment loader, K-contiguous storage.
// Lane l (r = l&15, half = l>>4) holds K = 8*half+0..7 (VGPR0-3) and
// K = 16+8*half+0..7 (VGPR4-7) of row r (M for A frags, N for B frags).
// With K contiguous this is exactly two 16-byte loads.
// ---------------------------------------------------------------------------
__device__ inline v16h load_frag_k1(const _Float16* p, int srow) {
  const int lane = threadIdx.x & 31;
  const int r = lane & 15, half = lane >> 4;
  const _Float16* q = p + (size_t)r * srow + 8 * half;
  const h8 lo = *(const h8*)q;          // K = 8*half + 0..7
  const h8 hi = *(const h8*)(q + 16);   // K = 16 + 8*half + 0..7
  v16h out;
#pragma unroll
  for (int i = 0; i < 8; ++i) { out[i] = lo[i]; out[8 + i] = hi[i]; }
  return out;
}

__device__ inline float rowmax16(float v) {
#pragma unroll
  for (int m = 8; m >= 1; m >>= 1) v = fmaxf(v, __shfl_xor(v, m, 16));
  return v;
}
__device__ inline float rowsum16(float v) {
#pragma unroll
  for (int m = 8; m >= 1; m >>= 1) v += __shfl_xor(v, m, 16);
  return v;
}

// ---------------------------------------------------------------------------
// Kernel 0: f32 -> f16 weight conversion (row-major kept: K=c contiguous)
// ---------------------------------------------------------------------------
__global__ void cvt_weights(const float* __restrict__ qw, const float* __restrict__ pw,
                            _Float16* __restrict__ qh, _Float16* __restrict__ ph) {
  const int i = blockIdx.x * blockDim.x + threadIdx.x;   // grid covers 768*256
  if (i < OC3 * C) qh[i] = (_Float16)qw[i];
  if (i < C * C)   ph[i] = (_Float16)pw[i];
}

// ---------------------------------------------------------------------------
// Kernel 1: GroupNorm -> f16 activations in [b][s][c] layout (c contiguous).
// ---------------------------------------------------------------------------
__global__ void groupnorm_kernel(const float* __restrict__ x,
                                 const float* __restrict__ gamma,
                                 const float* __restrict__ beta,
                                 _Float16* __restrict__ xn_t) {
  const int bg = blockIdx.x;
  const int b = bg >> 3, g = bg & 7;
  const int cpg = C / GRP;                       // 32
  const int nelem = cpg * HW;                    // 32768
  const float* xp = x + ((size_t)b * C + g * cpg) * HW;

  float s = 0.f, s2 = 0.f;
  for (int i = threadIdx.x; i < nelem; i += 256) {
    const float v = xp[i];
    s += v; s2 += v * v;
  }
  __shared__ float rs[256], rq[256];
  rs[threadIdx.x] = s; rq[threadIdx.x] = s2;
  __syncthreads();
  for (int st = 128; st > 0; st >>= 1) {
    if (threadIdx.x < st) { rs[threadIdx.x] += rs[threadIdx.x + st];
                            rq[threadIdx.x] += rq[threadIdx.x + st]; }
    __syncthreads();
  }
  const float mean = rs[0] / (float)nelem;
  const float var  = rq[0] / (float)nelem - mean * mean;
  const float inv  = rsqrtf(var + 1e-5f);

  for (int sp = threadIdx.x; sp < HW; sp += 256) {
    h8 buf[4];
#pragma unroll
    for (int cb = 0; cb < 4; ++cb)
#pragma unroll
      for (int ci = 0; ci < 8; ++ci) {
        const int c = cb * 8 + ci;
        const int gc = g * cpg + c;
        const float v = (xp[(size_t)c * HW + sp] - mean) * inv * gamma[gc] + beta[gc];
        buf[cb][ci] = (_Float16)v;
      }
    h8* dst = (h8*)(xn_t + ((size_t)b * HW + sp) * C + g * cpg);
    dst[0] = buf[0]; dst[1] = buf[1]; dst[2] = buf[2]; dst[3] = buf[3];
  }
}

// ---------------------------------------------------------------------------
// Kernel 2: QKV GEMM with 32x32 register tiles (2 M-frags x 2 N-frags,
// 4 accumulators -> 8 b128 loads feed 4 WMMAs per K-step: 2 loads/WMMA).
// acc[o][s] = sum_c W[o][c] * Xn[s][c] + bias[o];  M=768, N=1024, K=256.
// Output split at o=512 (== M-block 16 boundary, wave-uniform):
//   Q,K -> qk_t [b][s][512] (one 16B store per lane per tile)
//   V   -> v_h  [b][c][s]   (lane-coalesced rows)
// Grid: 8 * 24 * 32 waves / 8 = 768 blocks of 256 threads.
// ---------------------------------------------------------------------------
__global__ void gemm_qkv(const _Float16* __restrict__ W, const float* __restrict__ bias,
                         const _Float16* __restrict__ Xn, _Float16* __restrict__ qk_t,
                         _Float16* __restrict__ v_h) {
  const int g   = blockIdx.x * 8 + (threadIdx.x >> 5);
  const int b   = g / (24 * 32);
  const int rem = g % (24 * 32);
  const int mt2 = rem >> 5, nt2 = rem & 31;       // 32x32 output tile
  const _Float16* Wb = W + (size_t)mt2 * 32 * C;
  const _Float16* Xb = Xn + ((size_t)b * HW + nt2 * 32) * C;

  v8f acc00 = {}, acc01 = {}, acc10 = {}, acc11 = {};
#pragma unroll
  for (int kk = 0; kk < C; kk += 32) {
    const v16h a0 = load_frag_k1(Wb + kk, C);
    const v16h a1 = load_frag_k1(Wb + 16 * C + kk, C);
    const v16h b0 = load_frag_k1(Xb + kk, C);
    const v16h b1 = load_frag_k1(Xb + 16 * C + kk, C);
    acc00 = WMMA_F16(a0, b0, acc00);
    acc01 = WMMA_F16(a0, b1, acc01);
    acc10 = WMMA_F16(a1, b0, acc10);
    acc11 = WMMA_F16(a1, b1, acc11);
  }

  const int lane = threadIdx.x & 31, n = lane & 15, half = lane >> 4;
  const v8f* accs[4] = { &acc00, &acc01, &acc10, &acc11 };
#pragma unroll
  for (int t = 0; t < 4; ++t) {
    const int mt = mt2 * 2 + (t >> 1);
    const int nt = nt2 * 2 + (t & 1);
    const v8f& acc = *accs[t];
    h8 st;
#pragma unroll
    for (int j = 0; j < 8; ++j)
      st[j] = (_Float16)(acc[j] + bias[mt * 16 + 8 * half + j]);

    if (mt2 < 16) {       // wave-uniform: Q,K -> [s][o] layout
      *(h8*)(qk_t + ((size_t)b * HW + nt * 16 + n) * 512 + mt * 16 + 8 * half) = st;
    } else {              // V -> [c][s] layout
#pragma unroll
      for (int j = 0; j < 8; ++j) {
        const int c = mt * 16 + 8 * half + j - 512;
        v_h[((size_t)b * C + c) * HW + nt * 16 + n] = st[j];
      }
    }
  }
}

// ---------------------------------------------------------------------------
// Kernel 3: flash attention. One wave per (b, head, 16-query tile).
// scale = dh^-0.5 = 1/8. Per 32-key chunk: 4 score WMMAs, online softmax,
// P staged D-layout -> A-layout through per-wave LDS, 4 PV WMMAs.
// All operands K-contiguous (2 x b128 per fragment).
// Grid: 8*4*64 waves = 256 blocks of 256 threads.
// ---------------------------------------------------------------------------
__global__ void attn_kernel(const _Float16* __restrict__ qk_t,
                            const _Float16* __restrict__ v_h,
                            _Float16* __restrict__ at_t) {
  const int w  = threadIdx.x >> 5;
  const int g  = blockIdx.x * 8 + w;
  const int qt = g & 63, head = (g >> 6) & 3, b = g >> 8;

  const _Float16* qrow = qk_t + ((size_t)b * HW + qt * 16) * 512 + head * DH;
  const _Float16* krow = qk_t + (size_t)b * HW * 512 + C + head * DH;
  const _Float16* vrow = v_h + ((size_t)b * C + head * DH) * HW;

  __shared__ _Float16 pst[8][16 * 32];
  _Float16* pw = pst[w];
  const int lane = threadIdx.x & 31, n = lane & 15, half = lane >> 4;

  // Q fragments: A[m=d][k=c] = qrow[m*512 + k]
  const v16h qa0 = load_frag_k1(qrow, 512);
  const v16h qa1 = load_frag_k1(qrow + 32, 512);

  v8f o0 = {}, o1 = {}, o2 = {}, o3 = {};
  float rmax[8], rsum[8];
#pragma unroll
  for (int j = 0; j < 8; ++j) { rmax[j] = -1e30f; rsum[j] = 0.f; }
  const float scale = 0.125f;   // dh^-0.5

  for (int e0 = 0; e0 < HW; e0 += 32) {
    float p0[8], p1[8];
    {  // score tile 0: keys e0..e0+15; B[k=c][n=e] = krow[(e0+n)*512 + k]
      v8f s = {};
      s = WMMA_F16(qa0, load_frag_k1(krow + (size_t)e0 * 512, 512), s);
      s = WMMA_F16(qa1, load_frag_k1(krow + (size_t)e0 * 512 + 32, 512), s);
#pragma unroll
      for (int j = 0; j < 8; ++j) p0[j] = s[j] * scale;
    }
    {  // score tile 1: keys e0+16..e0+31
      v8f s = {};
      s = WMMA_F16(qa0, load_frag_k1(krow + (size_t)(e0 + 16) * 512, 512), s);
      s = WMMA_F16(qa1, load_frag_k1(krow + (size_t)(e0 + 16) * 512 + 32, 512), s);
#pragma unroll
      for (int j = 0; j < 8; ++j) p1[j] = s[j] * scale;
    }
    // online softmax (row-wise across 16 lanes per half-wave)
#pragma unroll
    for (int j = 0; j < 8; ++j) {
      const float tmax = fmaxf(rowmax16(p0[j]), rowmax16(p1[j]));
      const float nm = fmaxf(rmax[j], tmax);
      const float f  = __expf(rmax[j] - nm);
      rmax[j] = nm;
      p0[j] = __expf(p0[j] - nm);
      p1[j] = __expf(p1[j] - nm);
      rsum[j] = rsum[j] * f + rowsum16(p0[j]) + rowsum16(p1[j]);
      o0[j] *= f; o1[j] *= f; o2[j] *= f; o3[j] *= f;
      const int m = j + 8 * half;
      pw[m * 32 + n]      = (_Float16)p0[j];
      pw[m * 32 + 16 + n] = (_Float16)p1[j];
    }
    // P (16x32) @ V^T: B[k=e][n=c] = vrow[(cc*16+n)*HW + e0+k]
    const v16h pa = load_frag_k1(pw, 32);
    o0 = WMMA_F16(pa, load_frag_k1(vrow + 0 * 16 * HW + e0, HW), o0);
    o1 = WMMA_F16(pa, load_frag_k1(vrow + 1 * 16 * HW + e0, HW), o1);
    o2 = WMMA_F16(pa, load_frag_k1(vrow + 2 * 16 * HW + e0, HW), o2);
    o3 = WMMA_F16(pa, load_frag_k1(vrow + 3 * 16 * HW + e0, HW), o3);
  }

  // store out[d][c] into [b][s][c] layout (lane-coalesced 32B rows)
  _Float16* op = at_t + ((size_t)b * HW + qt * 16) * C + head * DH;
  const v8f* oacc[4] = { &o0, &o1, &o2, &o3 };
#pragma unroll
  for (int cc = 0; cc < 4; ++cc) {
    const v8f& o = *oacc[cc];
#pragma unroll
    for (int j = 0; j < 8; ++j) {
      const int m = j + 8 * half;                 // local query row (spatial)
      op[(size_t)m * C + cc * 16 + n] = (_Float16)(o[j] / rsum[j]);
    }
  }
}

// ---------------------------------------------------------------------------
// Kernel 4: proj GEMM (32x32 register tiles) + bias + residual, f32 out.
// M=256, N=1024, K=256 per batch. Grid: 8*8*32/8 = 256 blocks.
// ---------------------------------------------------------------------------
__global__ void gemm_proj(const _Float16* __restrict__ W, const float* __restrict__ bias,
                          const _Float16* __restrict__ At, const float* __restrict__ xres,
                          float* __restrict__ out) {
  const int g   = blockIdx.x * 8 + (threadIdx.x >> 5);
  const int nt2 = g & 31, mt2 = (g >> 5) & 7, b = g >> 8;
  const _Float16* Wb = W + (size_t)mt2 * 32 * C;
  const _Float16* Xb = At + ((size_t)b * HW + nt2 * 32) * C;

  v8f acc00 = {}, acc01 = {}, acc10 = {}, acc11 = {};
#pragma unroll
  for (int kk = 0; kk < C; kk += 32) {
    const v16h a0 = load_frag_k1(Wb + kk, C);
    const v16h a1 = load_frag_k1(Wb + 16 * C + kk, C);
    const v16h b0 = load_frag_k1(Xb + kk, C);
    const v16h b1 = load_frag_k1(Xb + 16 * C + kk, C);
    acc00 = WMMA_F16(a0, b0, acc00);
    acc01 = WMMA_F16(a0, b1, acc01);
    acc10 = WMMA_F16(a1, b0, acc10);
    acc11 = WMMA_F16(a1, b1, acc11);
  }

  const int lane = threadIdx.x & 31, n = lane & 15, half = lane >> 4;
  const v8f* accs[4] = { &acc00, &acc01, &acc10, &acc11 };
#pragma unroll
  for (int t = 0; t < 4; ++t) {
    const int mt = mt2 * 2 + (t >> 1);
    const int nt = nt2 * 2 + (t & 1);
    const v8f& acc = *accs[t];
#pragma unroll
    for (int j = 0; j < 8; ++j) {
      const int o = mt * 16 + 8 * half + j;
      const size_t idx = (size_t)b * C * HW + (size_t)o * HW + nt * 16 + n;
      out[idx] = xres[idx] + acc[j] + bias[o];
    }
  }
}

// ---------------------------------------------------------------------------
extern "C" void kernel_launch(void* const* d_in, const int* in_sizes, int n_in,
                              void* d_out, int out_size, void* d_ws, size_t ws_size,
                              hipStream_t stream) {
  const float* x        = (const float*)d_in[0];
  const float* gn_gamma = (const float*)d_in[1];
  const float* gn_beta  = (const float*)d_in[2];
  const float* qkv_w    = (const float*)d_in[3];
  const float* qkv_b    = (const float*)d_in[4];
  const float* proj_w   = (const float*)d_in[5];
  const float* proj_b   = (const float*)d_in[6];
  float* out = (float*)d_out;

  // workspace layout (f16, ~20.5 MB total -> resident in 192 MB L2)
  char* ws = (char*)d_ws;
  _Float16* qkv_wh  = (_Float16*)(ws);                 // 768*256       (384 KB)
  _Float16* proj_wh = (_Float16*)(ws + 393216);        // 256*256       (128 KB)
  _Float16* xn_t    = (_Float16*)(ws + 524288);        // [b][s][c]     (4 MB)
  _Float16* qk_t    = (_Float16*)(ws + 4718592);       // [b][s][512]   (8 MB)
  _Float16* v_h     = (_Float16*)(ws + 13107200);      // [b][c][s]     (4 MB)
  _Float16* at_t    = (_Float16*)(ws + 17301504);      // [b][s][c]     (4 MB)

  cvt_weights<<<768, 256, 0, stream>>>(qkv_w, proj_w, qkv_wh, proj_wh);
  groupnorm_kernel<<<B * GRP, 256, 0, stream>>>(x, gn_gamma, gn_beta, xn_t);
  gemm_qkv<<<(B * 24 * 32) / 8, 256, 0, stream>>>(qkv_wh, qkv_b, xn_t, qk_t, v_h);
  attn_kernel<<<(B * NH * 64) / 8, 256, 0, stream>>>(qk_t, v_h, at_t);
  gemm_proj<<<(B * 8 * 32) / 8, 256, 0, stream>>>(proj_wh, proj_b, at_t, x, out);
}
